// LSTM_27401891348777
// MI455X (gfx1250) — matrix operand order repaired and checked by reading, not compiled, not run on hardware
//
#include <hip/hip_runtime.h>
#include <hip/hip_bf16.h>

typedef __bf16 bf16;
typedef __attribute__((ext_vector_type(8)))  __bf16 v8bf;
typedef __attribute__((ext_vector_type(16))) __bf16 v16bf;
typedef __attribute__((ext_vector_type(8)))  float  v8f;

#define BB   64      // batch
#define TT   512     // time steps
#define DD   512     // input dim
#define UU   1024    // hidden dim
#define GG   4096    // 4*U (gates)
#define KTOT 1536    // D + U
#define DKT  16      // DD/32  (x-part k-tiles)
#define HKT  32      // UU/32  (h-part k-tiles)
#define NBLK 64      // persistent blocks (one per 16-wide u column group)

// ---------- one-time prep kernels ----------

__global__ void lstm_cvt_x(const float* __restrict__ x, bf16* __restrict__ xb, int n) {
    int i = blockIdx.x * 256 + threadIdx.x;
    if (i < n) xb[i] = (bf16)x[i];
}

// Pack [W (D x 4U); R (U x 4U)] fp32 into bf16 WMMA B-operand tiles.
// Tile (kt,nt) covers K rows [kt*32, kt*32+32), N cols [nt*16, nt*16+16).
// Within a tile (512 halfs): half index = chunk*256 + lane*8 + e, where the
// v16bf element e16 = chunk*8+e maps to k = chunk*16 + (lane/16)*8 + e, n = lane%16.
__global__ void lstm_pack_w(const float* __restrict__ W, const float* __restrict__ R,
                            bf16* __restrict__ Bpk) {
    int i = blockIdx.x * 256 + threadIdx.x;
    if (i >= KTOT * GG) return;
    int tile  = i >> 9;
    int rem   = i & 511;
    int kt    = tile >> 8;
    int nt    = tile & 255;
    int chunk = rem >> 8;
    int lane  = (rem >> 3) & 31;
    int e     = rem & 7;
    int e16   = chunk * 8 + e;
    int k = kt * 32 + chunk * 16 + (lane >> 4) * 8 + e;
    int n = nt * 16 + (lane & 15);
    float v = (k < DD) ? W[(size_t)k * GG + n] : R[(size_t)(k - DD) * GG + n];
    Bpk[i] = (bf16)v;
}

__global__ void lstm_init_state(bf16* __restrict__ h0, unsigned* __restrict__ bar) {
    int i = blockIdx.x * 256 + threadIdx.x;
    if (i < BB * UU) h0[i] = (bf16)0.0f;
    if (i == 0) *bar = 0u;
}

// ---------- persistent LSTM kernel ----------

__device__ __forceinline__ v16bf ldA(const bf16* p) {
    // A-operand 16-bit 16x32 layout: this lane needs 8 halfs at k-base and
    // 8 halfs at k-base+16 (k-base already includes (lane/16)*8).
    v8bf lo = *(const v8bf*)p;
    v8bf hi = *(const v8bf*)(p + 16);
    return __builtin_shufflevector(lo, hi, 0,1,2,3,4,5,6,7,8,9,10,11,12,13,14,15);
}

__device__ __forceinline__ v16bf ldB(const bf16* tb) {
    v8bf lo = *(const v8bf*)tb;
    v8bf hi = *(const v8bf*)(tb + 256);
    return __builtin_shufflevector(lo, hi, 0,1,2,3,4,5,6,7,8,9,10,11,12,13,14,15);
}

__device__ __forceinline__ float sigf(float x) { return 1.0f / (1.0f + __expf(-x)); }
// tanh(x) = 2*sigmoid(2x) - 1  (single v_exp_f32 instead of libm tanhf)
__device__ __forceinline__ float tanh_fast(float x) {
    return 2.0f / (1.0f + __expf(-2.0f * x)) - 1.0f;
}

// Device-wide sense-free barrier: monotonically increasing arrival counter.
__device__ __forceinline__ void grid_barrier(unsigned* bar, unsigned target, int tid) {
    __threadfence();           // publish this thread's h stores at agent scope
    __syncthreads();           // all waves in block arrived
    if (tid == 0) {
        __hip_atomic_fetch_add(bar, 1u, __ATOMIC_ACQ_REL, __HIP_MEMORY_SCOPE_AGENT);
        while (__hip_atomic_load(bar, __ATOMIC_ACQUIRE, __HIP_MEMORY_SCOPE_AGENT) < target) {
            __builtin_amdgcn_s_sleep(1);
        }
    }
    __syncthreads();           // release block
    __threadfence();           // invalidate stale near caches before reading h
}

__global__ __launch_bounds__(128)
void lstm_persistent(const bf16* __restrict__ xb,    // [B,T,D] bf16
                     const bf16* __restrict__ Bpk,   // packed [W;R] bf16
                     const float* __restrict__ bias, // [4U]
                     bf16* __restrict__ hbuf0,       // [B,U] bf16 (ping)
                     bf16* __restrict__ hbuf1,       // [B,U] bf16 (pong)
                     float* __restrict__ hf_out,     // [B,U] fp32 final output
                     unsigned* __restrict__ bar)     // grid barrier counter
{
    __shared__ float zsh[4][64][16];   // gate pre-activations, 16 KB
    __shared__ float csh[64][16];      // persistent cell state slice, 4 KB

    const int tid   = threadIdx.x;
    const int lane  = tid & 31;
    const int gate  = tid >> 5;                // wave -> gate (i,f,c,o)
    const int u0    = blockIdx.x * 16;         // column group within U
    const int ncol  = lane & 15;
    const int khalf = lane >> 4;

    // zero the cell state slice
    #pragma unroll
    for (int j = 0; j < 8; ++j) {
        const int idx = tid + 128 * j;
        csh[idx >> 4][idx & 15] = 0.0f;
    }
    __syncthreads();

    const float bv = bias[gate * UU + u0 + ncol];
    const int   nt = gate * (UU / 16) + blockIdx.x;

    const int m = lane & 15;
    const bf16* xrow0 = xb  + (size_t)m * TT * DD + khalf * 8;   // + t*DD per step
    const bf16* brow  = Bpk + (size_t)nt * 512    + (size_t)lane * 8;

    const size_t xrs = (size_t)TT * DD * 16;   // 16 batch rows per m-tile
    const size_t hrs = (size_t)UU * 16;
    const size_t bks = (size_t)256 * 512;      // halfs per k-tile of B

    for (int t = 0; t < TT; ++t) {
        const bf16* hin  = (t & 1) ? hbuf1 : hbuf0;
        bf16*       hout = (t & 1) ? hbuf0 : hbuf1;

        v8f acc0 = {bv,bv,bv,bv,bv,bv,bv,bv};
        v8f acc1 = acc0, acc2 = acc0, acc3 = acc0;

        const bf16* xrow = xrow0 + (size_t)t * DD;
        const bf16* hrow = hin + (size_t)m * UU + khalf * 8;

        // ---- K part 1: x_t @ W ----
        #pragma unroll 4
        for (int kt = 0; kt < DKT; ++kt) {
            v16bf bm = ldB(brow + (size_t)kt * bks);
            const bf16* p = xrow + kt * 32;
            v16bf a0 = ldA(p);
            v16bf a1 = ldA(p + xrs);
            v16bf a2 = ldA(p + 2 * xrs);
            v16bf a3 = ldA(p + 3 * xrs);
            acc0 = __builtin_amdgcn_wmma_f32_16x16x32_bf16(false, a0, false, bm,
                                                           (short)0, acc0, false, false);
            acc1 = __builtin_amdgcn_wmma_f32_16x16x32_bf16(false, a1, false, bm,
                                                           (short)0, acc1, false, false);
            acc2 = __builtin_amdgcn_wmma_f32_16x16x32_bf16(false, a2, false, bm,
                                                           (short)0, acc2, false, false);
            acc3 = __builtin_amdgcn_wmma_f32_16x16x32_bf16(false, a3, false, bm,
                                                           (short)0, acc3, false, false);
        }

        // ---- K part 2: h_{t-1} @ R ----
        #pragma unroll 4
        for (int kt = 0; kt < HKT; ++kt) {
            v16bf bm = ldB(brow + (size_t)(DKT + kt) * bks);
            const bf16* p = hrow + kt * 32;
            v16bf a0 = ldA(p);
            v16bf a1 = ldA(p + hrs);
            v16bf a2 = ldA(p + 2 * hrs);
            v16bf a3 = ldA(p + 3 * hrs);
            acc0 = __builtin_amdgcn_wmma_f32_16x16x32_bf16(false, a0, false, bm,
                                                           (short)0, acc0, false, false);
            acc1 = __builtin_amdgcn_wmma_f32_16x16x32_bf16(false, a1, false, bm,
                                                           (short)0, acc1, false, false);
            acc2 = __builtin_amdgcn_wmma_f32_16x16x32_bf16(false, a2, false, bm,
                                                           (short)0, acc2, false, false);
            acc3 = __builtin_amdgcn_wmma_f32_16x16x32_bf16(false, a3, false, bm,
                                                           (short)0, acc3, false, false);
        }

        // C/D layout: VGPR r holds m_in_tile = r + 8*(lane/16), col = lane%16
        #pragma unroll
        for (int r = 0; r < 8; ++r) {
            const int mi = 8 * khalf + r;
            zsh[gate][ 0 + mi][ncol] = acc0[r];
            zsh[gate][16 + mi][ncol] = acc1[r];
            zsh[gate][32 + mi][ncol] = acc2[r];
            zsh[gate][48 + mi][ncol] = acc3[r];
        }
        __syncthreads();

        // Elementwise recurrence: 64x16 outputs, 128 threads -> 8 each
        #pragma unroll
        for (int j = 0; j < 8; ++j) {
            const int idx  = tid + 128 * j;    // 0..1023
            const int mrow = idx >> 4;
            const int n    = idx & 15;
            const float zi = zsh[0][mrow][n];
            const float zf = zsh[1][mrow][n];
            const float zc = zsh[2][mrow][n];
            const float zo = zsh[3][mrow][n];
            const float cn = sigf(zf) * csh[mrow][n] + sigf(zi) * tanh_fast(zc);
            csh[mrow][n] = cn;
            const float hn = sigf(zo) * tanh_fast(cn);
            const size_t gi = (size_t)mrow * UU + u0 + n;
            hout[gi] = (bf16)hn;
            if (t == TT - 1) hf_out[gi] = hn;
        }

        // Device-wide barrier: h_t fully published before any block reads it
        grid_barrier(bar, (unsigned)(NBLK * (t + 1)), tid);
    }
}

// ---------- host launcher ----------

extern "C" void kernel_launch(void* const* d_in, const int* in_sizes, int n_in,
                              void* d_out, int out_size, void* d_ws, size_t ws_size,
                              hipStream_t stream) {
    const float* x    = (const float*)d_in[0];   // [B,T,D]
    const float* W    = (const float*)d_in[1];   // [D,4U]
    const float* R    = (const float*)d_in[2];   // [U,4U]
    const float* bias = (const float*)d_in[3];   // [4U]
    float* out = (float*)d_out;                  // [B,U]

    char* ws = (char*)d_ws;
    size_t off = 0;
    auto carve = [&](size_t bytes) -> char* {
        char* p = ws + off;
        off += (bytes + 255) & ~(size_t)255;
        return p;
    };
    bf16*     xbuf = (bf16*)    carve((size_t)BB * TT * DD * 2);   // 32 MB
    bf16*     Bpk  = (bf16*)    carve((size_t)KTOT * GG * 2);      // 12.6 MB
    bf16*     hA   = (bf16*)    carve((size_t)BB * UU * 2);
    bf16*     hB   = (bf16*)    carve((size_t)BB * UU * 2);
    unsigned* bar  = (unsigned*)carve(256);

    const int nX = BB * TT * DD;
    lstm_cvt_x     <<<(nX + 255) / 256, 256, 0, stream>>>(x, xbuf, nX);
    lstm_pack_w    <<<(KTOT * GG + 255) / 256, 256, 0, stream>>>(W, R, Bpk);
    lstm_init_state<<<(BB * UU + 255) / 256, 256, 0, stream>>>(hA, bar);

    lstm_persistent<<<NBLK, 128, 0, stream>>>(xbuf, Bpk, bias, hA, hB, out, bar);
}
